// NONLocalBlock2D_13709535609475
// MI455X (gfx1250) — matrix operand order repaired and checked by reading, not compiled, not run on hardware
//
#include <hip/hip_runtime.h>
#include <hip/hip_bf16.h>
#include <stdint.h>

#define BB 8
#define CC 512
#define NN 4096          // 64*64
#define INTER 32

typedef __attribute__((ext_vector_type(16))) _Float16 v16h;
typedef __attribute__((ext_vector_type(8)))  float    v8f;
typedef __attribute__((ext_vector_type(4)))  unsigned int u32x4;
typedef __attribute__((ext_vector_type(8)))  int          i32x8;
typedef __attribute__((ext_vector_type(4)))  int          i32x4;

union Vh {
    v16h v;
    _Float16 h[16];
    unsigned int u[8];
};

// ---------------------------------------------------------------------------
// Kernel 1: projections.  For each 16-row n-tile: theta = x^T W_theta^T,
// g = x^T W_g^T, theta_c = theta - rowmean(theta).  WMMA f32_16x16x32_f16,
// K loop over C in chunks of 32 (16 iterations, 4 WMMAs each).
// ---------------------------------------------------------------------------
__global__ __launch_bounds__(256)
void proj_kernel(const float* __restrict__ x,       // [B, C, N]
                 const float* __restrict__ w_g,     // [INTER, C]
                 const float* __restrict__ w_theta, // [INTER, C]
                 _Float16* __restrict__ theta,      // [B*N, 32]
                 _Float16* __restrict__ thetac,     // [B*N, 32]
                 _Float16* __restrict__ gx)         // [B*N, 32]
{
    const int lane = threadIdx.x & 31;
    const int wave = threadIdx.x >> 5;
    const int tile = blockIdx.x * 8 + wave;     // 0 .. B*N/16-1
    const int b    = tile >> 8;                 // tile / (N/16)
    const int n0   = (tile & 255) << 4;
    const int half = lane >> 4;                 // 0 | 1
    const int lid  = lane & 15;

    const float* xb = x + (size_t)b * CC * NN;

    v8f t0 = {}, t1 = {}, g0 = {}, g1 = {};

    for (int kc = 0; kc < CC; kc += 32) {
        // A tile: x^T rows n0..n0+15, K = channels kc..kc+31 (16-bit A layout)
        Vh a;
#pragma unroll
        for (int v = 0; v < 8; ++v) {
            const int K0 = ((v < 4) ? 2 * v : 16 + 2 * (v - 4)) + (half ? 8 : 0);
            a.h[2 * v]     = (_Float16)xb[(size_t)(kc + K0)     * NN + n0 + lid];
            a.h[2 * v + 1] = (_Float16)xb[(size_t)(kc + K0 + 1) * NN + n0 + lid];
        }
        // B tiles: W^T blocks [K=32 channels x 16 outputs]; channel pairs
        // contiguous -> float2 loads.
        Vh bt0, bt1, bg0, bg1;
#pragma unroll
        for (int v = 0; v < 8; ++v) {
            const int K0 = ((v < 4) ? 2 * v : 16 + 2 * (v - 4)) + (half ? 8 : 0);
            const float2* wt_lo = (const float2*)(w_theta + (size_t)(lid)      * CC + kc);
            const float2* wt_hi = (const float2*)(w_theta + (size_t)(lid + 16) * CC + kc);
            const float2* wg_lo = (const float2*)(w_g     + (size_t)(lid)      * CC + kc);
            const float2* wg_hi = (const float2*)(w_g     + (size_t)(lid + 16) * CC + kc);
            float2 a0 = wt_lo[K0 >> 1], a1 = wt_hi[K0 >> 1];
            float2 c0 = wg_lo[K0 >> 1], c1 = wg_hi[K0 >> 1];
            bt0.h[2 * v] = (_Float16)a0.x; bt0.h[2 * v + 1] = (_Float16)a0.y;
            bt1.h[2 * v] = (_Float16)a1.x; bt1.h[2 * v + 1] = (_Float16)a1.y;
            bg0.h[2 * v] = (_Float16)c0.x; bg0.h[2 * v + 1] = (_Float16)c0.y;
            bg1.h[2 * v] = (_Float16)c1.x; bg1.h[2 * v + 1] = (_Float16)c1.y;
        }
        t0 = __builtin_amdgcn_wmma_f32_16x16x32_f16(false, a.v, false, bt0.v, (short)0, t0, false, false);
        t1 = __builtin_amdgcn_wmma_f32_16x16x32_f16(false, a.v, false, bt1.v, (short)0, t1, false, false);
        g0 = __builtin_amdgcn_wmma_f32_16x16x32_f16(false, a.v, false, bg0.v, (short)0, g0, false, false);
        g1 = __builtin_amdgcn_wmma_f32_16x16x32_f16(false, a.v, false, bg1.v, (short)0, g1, false, false);
    }

#pragma unroll
    for (int r = 0; r < 8; ++r) {
        float rs = t0[r] + t1[r];
        rs += __shfl_xor(rs, 1);
        rs += __shfl_xor(rs, 2);
        rs += __shfl_xor(rs, 4);
        rs += __shfl_xor(rs, 8);
        const float mean = rs * (1.0f / 32.0f);

        const int    row  = n0 + r + 8 * half;           // D layout: M = r + 8*(lane/16)
        const size_t base = ((size_t)b * NN + row) * 32;
        theta [base + lid]      = (_Float16)t0[r];
        theta [base + lid + 16] = (_Float16)t1[r];
        thetac[base + lid]      = (_Float16)(t0[r] - mean);
        thetac[base + lid + 16] = (_Float16)(t1[r] - mean);
        gx    [base + lid]      = (_Float16)g0[r];
        gx    [base + lid + 16] = (_Float16)g1[r];
    }
}

// ---------------------------------------------------------------------------
// Tensor Data Mover: DMA one 32x32 f16 tile (row stride 32 elements) from
// global memory into LDS.  2D descriptor, data_size=2B, tracked by TENSORcnt.
// ---------------------------------------------------------------------------
__device__ __forceinline__ void tdm_load_tile32x32(const _Float16* gsrc,
                                                   unsigned int lds_off)
{
    const unsigned long long ga = (unsigned long long)(uintptr_t)gsrc;
    u32x4 g0;
    g0[0] = 1u;                                     // count=1, user descriptor
    g0[1] = lds_off;                                // lds_addr
    g0[2] = (unsigned int)(ga & 0xFFFFFFFFu);       // global_addr[31:0]
    g0[3] = (unsigned int)((ga >> 32) & 0x01FFFFFFu) | 0x80000000u; // addr[56:32] | type=2

    i32x8 g1;
    g1[0] = 0x10000;        // workgroup_mask=0, data_size=1 (2 bytes)
    g1[1] = 32 << 16;       // tensor_dim0 = 32 (elements per row)
    g1[2] = 32 << 16;       // tensor_dim1 = 32 (rows)
    g1[3] = 32 << 16;       // tile_dim0 = 32
    g1[4] = 32;             // tile_dim1 = 32, tile_dim2 = 0
    g1[5] = 32;             // tensor_dim0_stride = 32
    g1[6] = 0;
    g1[7] = 0;

    const i32x4 z4 = {0, 0, 0, 0};
#if __clang_major__ >= 23
    const i32x8 z8 = {0, 0, 0, 0, 0, 0, 0, 0};
    __builtin_amdgcn_tensor_load_to_lds(g0, g1, z4, z4, z8, 0);
#else
    __builtin_amdgcn_tensor_load_to_lds(g0, g1, z4, z4, 0);
#endif
}

// ---------------------------------------------------------------------------
// Kernel 2: fused flash attention.  One wave owns a 16-query tile; the 8
// waves of a block share the same batch and key schedule, so key/value
// blocks are staged once per block into LDS by the Tensor Data Mover
// (double-buffered, TENSORcnt-synchronized) instead of 8x-redundant global
// loads.  S^T = theta_k (A) x theta_c_q^T (B): each lane owns one query
// column -> scalar online-softmax state; P^T re-packed into B-layout with
// 8 shfl_xor(16) swaps; y^T += g^T (A) x P^T (B).  4 WMMAs / 32-key block.
// ---------------------------------------------------------------------------
__global__ __launch_bounds__(256)
void attn_kernel(const _Float16* __restrict__ theta,   // keys   [B*N,32]
                 const _Float16* __restrict__ thetac,  // queries[B*N,32]
                 const _Float16* __restrict__ gx,      // values [B*N,32]
                 float* __restrict__ y)                // [B*N,32]
{
    __shared__ _Float16 sk[2][32 * 32];   // staged keys   (2 KB per buffer)
    __shared__ _Float16 sg[2][32 * 32];   // staged values (2 KB per buffer)

    const int lane = threadIdx.x & 31;
    const int wave = threadIdx.x >> 5;
    const int tile = blockIdx.x * 8 + wave;
    const int b    = tile >> 8;            // uniform across the block
    const int q0   = (tile & 255) << 4;
    const int half = lane >> 4;
    const int lid  = lane & 15;

    const _Float16* thb = theta + (size_t)b * NN * 32;
    const _Float16* gb  = gx    + (size_t)b * NN * 32;

    // Q in B-matrix layout: lane -> query column lid; K pairs contiguous.
    Vh bq;
    {
        const unsigned int* qrow =
            (const unsigned int*)(thetac + ((size_t)b * NN + q0 + lid) * 32 + (half ? 16 : 0));
#pragma unroll
        for (int v = 0; v < 8; ++v) bq.u[v] = qrow[v];
    }

    float m = -3.0e38f, s = 0.0f;
    v8f y0 = {}, y1 = {};
    const v8f cz = {};
    const float scale = 1.0f / 32.0f;

    // Prefill buffer 0 (keys+values for j=0) via TDM from wave 0.
    if (wave == 0) {
        tdm_load_tile32x32(thb, (unsigned int)(uintptr_t)&sk[0][0]);
        tdm_load_tile32x32(gb,  (unsigned int)(uintptr_t)&sg[0][0]);
    }

    for (int jb = 0; jb < NN / 32; ++jb) {
        const int cur = jb & 1;
        if (wave == 0) __builtin_amdgcn_s_wait_tensorcnt((short)0);
        __syncthreads();                       // staged data visible to all waves
        if (wave == 0 && jb + 1 < NN / 32) {   // overlap DMA of next block
            const size_t j2 = (size_t)(jb + 1) * 32 * 32;
            tdm_load_tile32x32(thb + j2, (unsigned int)(uintptr_t)&sk[cur ^ 1][0]);
            tdm_load_tile32x32(gb + j2,  (unsigned int)(uintptr_t)&sg[cur ^ 1][0]);
        }
        const _Float16* skc = sk[cur];
        const _Float16* sgc = sg[cur];

        // Key tiles in A-matrix layout (rows = keys) from LDS, dword reads.
        Vh ak0, ak1;
        {
            const unsigned int* kr0 = (const unsigned int*)(skc + (size_t)lid * 32);
            const unsigned int* kr1 = (const unsigned int*)(skc + (size_t)(16 + lid) * 32);
#pragma unroll
            for (int v = 0; v < 8; ++v) {
                const int K0 = (half ? 16 : 0) + 2 * v;
                ak0.u[v] = kr0[K0 >> 1];
                ak1.u[v] = kr1[K0 >> 1];
            }
        }
        v8f s0 = __builtin_amdgcn_wmma_f32_16x16x32_f16(false, ak0.v, false, bq.v, (short)0, cz, false, false);
        v8f s1 = __builtin_amdgcn_wmma_f32_16x16x32_f16(false, ak1.v, false, bq.v, (short)0, cz, false, false);

        // Online softmax along keys; each lane holds 16 of the 32 scores of
        // its query column, the other 16 live in lane^16.
        float t0[8], t1[8];
        float mx = -3.0e38f;
#pragma unroll
        for (int r = 0; r < 8; ++r) {
            t0[r] = s0[r] * scale;
            t1[r] = s1[r] * scale;
            mx = fmaxf(mx, fmaxf(t0[r], t1[r]));
        }
        mx = fmaxf(mx, __shfl_xor(mx, 16));
        const float mn    = fmaxf(m, mx);
        const float alpha = __expf(m - mn);
        m = mn;

        float p0[8], p1[8], ls = 0.0f;
#pragma unroll
        for (int r = 0; r < 8; ++r) {
            p0[r] = __expf(t0[r] - mn);
            p1[r] = __expf(t1[r] - mn);
            ls += p0[r] + p1[r];
        }
        ls += __shfl_xor(ls, 16);
        s = s * alpha + ls;
#pragma unroll
        for (int r = 0; r < 8; ++r) { y0[r] *= alpha; y1[r] *= alpha; }

        // Re-pack P^T (two D-layout tiles) into a B-layout tile [32K x 16q]
        // with a half-wave register swap (no LDS round trip).
        float lowp[8], highp[8];
#pragma unroll
        for (int r = 0; r < 8; ++r) {
            const float own  = half ? p1[r] : p0[r];
            const float xfer = half ? p0[r] : p1[r];
            const float recv = __shfl_xor(xfer, 16);
            lowp[r]  = half ? recv : own;
            highp[r] = half ? own  : recv;
        }
        Vh bp;
#pragma unroll
        for (int v = 0; v < 8; ++v) {
            const float e0 = (2 * v < 8)     ? lowp[2 * v]     : highp[2 * v - 8];
            const float e1 = (2 * v + 1 < 8) ? lowp[2 * v + 1] : highp[2 * v + 1 - 8];
            bp.h[2 * v]     = (_Float16)e0;
            bp.h[2 * v + 1] = (_Float16)e1;
        }

        // g^T tiles in A-matrix layout (row = inter channel, K = keys) from LDS.
        Vh ag0, ag1;
#pragma unroll
        for (int v = 0; v < 8; ++v) {
            const int K0 = (half ? 16 : 0) + 2 * v;
            ag0.h[2 * v]     = sgc[(size_t)K0 * 32 + lid];
            ag0.h[2 * v + 1] = sgc[(size_t)(K0 + 1) * 32 + lid];
            ag1.h[2 * v]     = sgc[(size_t)K0 * 32 + 16 + lid];
            ag1.h[2 * v + 1] = sgc[(size_t)(K0 + 1) * 32 + 16 + lid];
        }
        y0 = __builtin_amdgcn_wmma_f32_16x16x32_f16(false, ag0.v, false, bp.v, (short)0, y0, false, false);
        y1 = __builtin_amdgcn_wmma_f32_16x16x32_f16(false, ag1.v, false, bp.v, (short)0, y1, false, false);

        __syncthreads();                     // done with cur before overwrite
    }

    // y^T D layout: lane -> query q0+lid, inter rows r + 8*half (+16 tile1).
    const float inv = 1.0f / s;
    float* yr = y + ((size_t)b * NN + q0 + lid) * 32;
#pragma unroll
    for (int r = 0; r < 8; ++r) {
        yr[r + 8 * half]      = y0[r] * inv;
        yr[16 + r + 8 * half] = y1[r] * inv;
    }
}

// ---------------------------------------------------------------------------
// Kernel 3: output projection + residual.  out[b,c,n] = x[b,c,n] +
// sum_o w_W[c,o] * y[b,n,o].  Memory-bound; float4 loads, y is L2-resident.
// ---------------------------------------------------------------------------
__global__ __launch_bounds__(256)
void out_kernel(const float* __restrict__ x,
                const float* __restrict__ wW,   // [C, INTER]
                const float* __restrict__ y,    // [B*N, 32]
                float* __restrict__ out)
{
    const size_t idx = (size_t)blockIdx.x * 256 + threadIdx.x;   // B*C*N total
    const int n = (int)(idx & (NN - 1));
    const int c = (int)((idx >> 12) & (CC - 1));
    const int b = (int)(idx >> 21);

    const float4* yr = (const float4*)(y + ((size_t)b * NN + n) * 32);
    const float4* wr = (const float4*)(wW + (size_t)c * 32);
    float acc = 0.0f;
#pragma unroll
    for (int k = 0; k < 8; ++k) {
        const float4 yv = yr[k];
        const float4 wv = wr[k];
        acc += yv.x * wv.x + yv.y * wv.y + yv.z * wv.z + yv.w * wv.w;
    }
    out[idx] = x[idx] + acc;
}

// ---------------------------------------------------------------------------
extern "C" void kernel_launch(void* const* d_in, const int* in_sizes, int n_in,
                              void* d_out, int out_size, void* d_ws, size_t ws_size,
                              hipStream_t stream) {
    const float* x       = (const float*)d_in[0];
    const float* w_g     = (const float*)d_in[1];
    const float* w_theta = (const float*)d_in[2];
    const float* w_W     = (const float*)d_in[3];
    float* out = (float*)d_out;

    char* ws = (char*)d_ws;
    const size_t nproj = (size_t)BB * NN * 32;          // 1,048,576 elems
    _Float16* theta  = (_Float16*)(ws);                 // 2 MB
    _Float16* thetac = (_Float16*)(ws + 2 * nproj);     // 2 MB
    _Float16* gx     = (_Float16*)(ws + 4 * nproj);     // 2 MB
    float*    yv     = (float*)   (ws + 6 * nproj);     // 4 MB

    proj_kernel<<<256, 256, 0, stream>>>(x, w_g, w_theta, theta, thetac, gx);
    attn_kernel<<<256, 256, 0, stream>>>(theta, thetac, gx, yv);
    out_kernel<<<(BB * CC * NN) / 256, 256, 0, stream>>>(x, w_W, yv, out);
}
